// HistogramLoss_36790689857604
// MI455X (gfx1250) — matrix operand order repaired and checked by reading, not compiled, not run on hardware
//
#include <hip/hip_runtime.h>

// ---------------------------------------------------------------------------
// HistogramLoss on MI455X (gfx1250, wave32, WMMA)
//  - split-f16 (hi+lo) Gram matrix via v_wmma_f32_16x16x32_f16
//  - D fixed at 128 (reference shape): K-loop fully unrolled, all fragment
//    addresses hoisted -> loads use immediate offsets only, so no VALU
//    address math inside the loop (kills the WMMA->VALU WAR v_nop padding)
//  - fused upper-triangular histogram scatter (per-wave LDS hists)
// ---------------------------------------------------------------------------

typedef __attribute__((ext_vector_type(16))) _Float16 v16h;
typedef __attribute__((ext_vector_type(8)))  _Float16 v8h;
typedef __attribute__((ext_vector_type(8)))  float    v8f;

#define NBINS     100
#define TILE      128          // 128x128 output tile per block (8 waves x 16 rows)
#define DK        128          // embedding dim (reference shape)
#define CLIP_EPS  1e-6f

__device__ __forceinline__ v8f wmma_f16(v16h a, v16h b, v8f c) {
    // (neg_a, A, neg_b, B, c_mod, C, reuse_a, reuse_b)
    return __builtin_amdgcn_wmma_f32_16x16x32_f16(false, a, false, b,
                                                  (short)0, c, false, false);
}

// A fragment (16x32 f16, MxK), lane l: M = l&15, half = l>>4.
// p already includes row*DK + 8*half; per-chunk offsets are immediates.
__device__ __forceinline__ v16h ldfragA(const _Float16* p) {
    v8h x = *(const v8h*)(p);        // K = kb .. kb+7
    v8h y = *(const v8h*)(p + 16);   // K = kb+16 .. kb+23
    v16h a;
#pragma unroll
    for (int e = 0; e < 8; ++e) { a[e] = x[e]; a[e + 8] = y[e]; }
    return a;
}

// B fragment (32x16 f16, KxN), lane l: N = l&15, half = l>>4.
// p already includes row*DK + 8*half.
__device__ __forceinline__ v16h ldfragB(const _Float16* p) {
    v8h x = *(const v8h*)(p);        // K = kb .. kb+7
    v8h y = *(const v8h*)(p + 16);   // K = kb+16 .. kb+23
    v16h b;
#pragma unroll
    for (int e = 0; e < 8; ++e) { b[e] = x[e]; b[e + 8] = y[e]; }
    return b;
}

// --------------------------- Kernel 1: normalize ---------------------------
__global__ void normalize_split_kernel(const float* __restrict__ emb,
                                       _Float16* __restrict__ enHi,
                                       _Float16* __restrict__ enLo, int D) {
    __shared__ float red[256];
    const int b = blockIdx.x;
    const int t = threadIdx.x;
    float x = emb[(size_t)b * D + t];
    red[t] = x * x;
    __syncthreads();
    for (int s = blockDim.x >> 1; s > 0; s >>= 1) {
        if (t < s) red[t] += red[t + s];
        __syncthreads();
    }
    const float inv = 1.0f / sqrtf(red[0]);
    const float xn = x * inv;
    const _Float16 hi = (_Float16)xn;
    const _Float16 lo = (_Float16)(xn - (float)hi);
    enHi[(size_t)b * D + t] = hi;
    enLo[(size_t)b * D + t] = lo;
}

// --------------------------- Kernel 2: zero hist ---------------------------
__global__ void zero_hist_kernel(float* __restrict__ g) {
    if (threadIdx.x < 2 * NBINS) g[threadIdx.x] = 0.0f;
}

// ------------------- Kernel 3: fused WMMA GEMM + histogram -----------------
__global__ __launch_bounds__(256)
void pairwise_hist_kernel(const _Float16* __restrict__ enHi,
                          const _Float16* __restrict__ enLo,
                          const int* __restrict__ classes,
                          float* __restrict__ gHist,
                          int T, float step, float invStep) {
    __shared__ float hloc[8 * 2 * NBINS];   // per-wave pos|neg histograms

    // Map linear block id to upper-triangular tile (tr, tc), tc >= tr.
    int bid = blockIdx.x;
    int tr = 0;
    while (bid >= T - tr) { bid -= T - tr; ++tr; }
    const int tc = tr + bid;

    const int tid  = threadIdx.x;
    for (int t = tid; t < 8 * 2 * NBINS; t += blockDim.x) hloc[t] = 0.0f;
    __syncthreads();

    const int wave = tid >> 5;
    const int lane = tid & 31;
    const int half = lane >> 4;
    const int l16  = lane & 15;

    // Hoist all fragment base addresses (loop-invariant; per-chunk offsets
    // inside the unrolled loop are pure instruction immediates).
    const int rowA = tr * TILE + wave * 16 + l16;
    const _Float16* aHiP = enHi + (size_t)rowA * DK + 8 * half;
    const _Float16* aLoP = enLo + (size_t)rowA * DK + 8 * half;
    const _Float16* bHiP[8];
    const _Float16* bLoP[8];
#pragma unroll
    for (int nt = 0; nt < 8; ++nt) {
        const int rowB = tc * TILE + nt * 16 + l16;
        bHiP[nt] = enHi + (size_t)rowB * DK + 8 * half;
        bLoP[nt] = enLo + (size_t)rowB * DK + 8 * half;
    }

    v8f acc[8] = {};                                // 16x128 strip of C

#pragma unroll
    for (int kc = 0; kc < DK / 32; ++kc) {
        const v16h aHi = ldfragA(aHiP + kc * 32);
        const v16h aLo = ldfragA(aLoP + kc * 32);
#pragma unroll
        for (int nt = 0; nt < 8; nt += 2) {
            const v16h bHi0 = ldfragB(bHiP[nt]     + kc * 32);
            const v16h bLo0 = ldfragB(bLoP[nt]     + kc * 32);
            const v16h bHi1 = ldfragB(bHiP[nt + 1] + kc * 32);
            const v16h bLo1 = ldfragB(bLoP[nt + 1] + kc * 32);
            // Two independent accumulator chains interleaved.
            acc[nt]     = wmma_f16(aHi, bHi0, acc[nt]);
            acc[nt + 1] = wmma_f16(aHi, bHi1, acc[nt + 1]);
            acc[nt]     = wmma_f16(aHi, bLo0, acc[nt]);
            acc[nt + 1] = wmma_f16(aHi, bLo1, acc[nt + 1]);
            acc[nt]     = wmma_f16(aLo, bHi0, acc[nt]);
            acc[nt + 1] = wmma_f16(aLo, bHi1, acc[nt + 1]);
        }
    }

    // C layout: acc[nt][r] is element (M = r + 8*half, N = l16) of tile nt.
    const int iBase = tr * TILE + wave * 16 + 8 * half;
    const int jBase = tc * TILE + l16;
    int clsI[8], clsJ[8];
#pragma unroll
    for (int r = 0; r < 8; ++r)  clsI[r]  = classes[iBase + r];
#pragma unroll
    for (int nt = 0; nt < 8; ++nt) clsJ[nt] = classes[jBase + nt * 16];

    float* const hw = hloc + wave * (2 * NBINS);    // this wave's hist pair

#pragma unroll
    for (int nt = 0; nt < 8; ++nt) {
        const int jj = jBase + nt * 16;
#pragma unroll
        for (int r = 0; r < 8; ++r) {
            const int i = iBase + r;
            if (jj > i) {                           // strict upper triangle
                float s = acc[nt][r];
                s = fminf(fmaxf(s, -1.0f + CLIP_EPS), 1.0f - CLIP_EPS);
                int bj = (int)floorf((s + 1.0f - CLIP_EPS) * invStep);
                bj = max(0, min(bj, NBINS - 2));
                const float tj  = (float)bj * step - 1.0f;
                const float whi = (s - tj) * invStep;
                const float wlo = (tj + step - s) * invStep;
                float* h = (clsI[r] == clsJ[nt]) ? hw : (hw + NBINS);
                atomicAdd(&h[bj],     wlo);
                atomicAdd(&h[bj + 1], whi);
            }
        }
    }
    __syncthreads();

    // Reduce 8 wave-hists and push to global.
    for (int t = tid; t < 2 * NBINS; t += blockDim.x) {
        float s = 0.0f;
#pragma unroll
        for (int w = 0; w < 8; ++w) s += hloc[w * (2 * NBINS) + t];
        atomicAdd(&gHist[t], s);
    }
}

// --------------------------- Kernel 4: finalize ----------------------------
__global__ void finalize_kernel(const float* __restrict__ g,
                                float* __restrict__ out) {
    if (threadIdx.x == 0) {
        float sp = 0.0f, sn = 0.0f;
        for (int k = 0; k < NBINS; ++k) { sp += g[k]; sn += g[NBINS + k]; }
        float cdf = 0.0f, loss = 0.0f;
        for (int k = 0; k < NBINS; ++k) {
            cdf  += g[k] / sp;                       // cumsum of hist_pos
            loss += (g[NBINS + k] / sn) * cdf;       // dot with hist_neg
        }
        out[0] = loss;
    }
}

// ---------------------------------------------------------------------------
extern "C" void kernel_launch(void* const* d_in, const int* in_sizes, int n_in,
                              void* d_out, int out_size, void* d_ws, size_t ws_size,
                              hipStream_t stream) {
    const float* emb     = (const float*)d_in[0];   // [B, D] f32
    const int*   classes = (const int*)d_in[1];     // [B] i32
    const int Bn = in_sizes[1];
    const int Dn = in_sizes[0] / Bn;                // 4096, 128 (DK)

    char* ws = (char*)d_ws;
    _Float16* enHi = (_Float16*)ws;                         // B*D f16
    _Float16* enLo = enHi + (size_t)Bn * Dn;                // B*D f16
    float*    gHist = (float*)(enLo + (size_t)Bn * Dn);     // 200 f32
    float*    out   = (float*)d_out;

    const float step    = 2.0f / (float)(NBINS - 1);
    const float invStep = (float)(NBINS - 1) / 2.0f;

    normalize_split_kernel<<<Bn, Dn, 0, stream>>>(emb, enHi, enLo, Dn);
    zero_hist_kernel<<<1, 256, 0, stream>>>(gHist);

    const int T = Bn / TILE;                        // 32 tiles per dim
    const int nblocks = T * (T + 1) / 2;            // 528 upper-tri tiles
    pairwise_hist_kernel<<<nblocks, 256, 0, stream>>>(enHi, enLo, classes,
                                                      gHist, T, step, invStep);

    finalize_kernel<<<1, 32, 0, stream>>>(gHist, out);
}
